// EBEPMLPModelWrapper_5514738008157
// MI455X (gfx1250) — matrix operand order repaired and checked by reading, not compiled, not run on hardware
//
#include <hip/hip_runtime.h>

// ---------------------------------------------------------------------------
// EB-EP MLP fixed-point energy model, MI455X (gfx1250, wave32, WMMA + async LDS)
//
//   Afull = x @ W0^T + b0                       (one-time, f32 WMMA)
//   repeat 100x (simultaneous update):
//     h1' = 0.5*h1 + 0.5*(Afull + h2 @ W1)      (f16 WMMA, f32 accum)
//     h2' = 0.5*h2 + 0.5*(h1 @ W1^T + b1)
//   out[b] = 0.5(|h1|^2+|h2|^2) - x.vb - <h1,Afull> - <h2, h1@W1^T + b1>
//
// ~429 GFLOP of square GEMMs over a ~32MB working set (L2-resident, 192MB)
// => compute bound => v_wmma_f32_16x16x32_f16 for the iteration GEMMs.
// Staging uses GLOBAL_LOAD_ASYNC_TO_LDS_B128 (ASYNCcnt) with LDS double
// buffering so the next K-chunk streams in while WMMAs consume the current.
// ---------------------------------------------------------------------------

typedef __attribute__((ext_vector_type(16))) _Float16 v16h;
typedef __attribute__((ext_vector_type(8)))  float    v8f;
typedef __attribute__((ext_vector_type(2)))  float    v2f;

#define BATCH 1024
#define DIN   784
#define HD    1024   // H1 == H2 == 1024
#define TSTEPS 100

// --- CDNA5 async memory->LDS helpers (inline asm; see CDNA5 ISA) --------
__device__ __forceinline__ void async_ldst_b128(void* lds_ptr, const void* gptr) {
    // VDST vgpr = LDS byte offset (low 32 bits of flat LDS address),
    // VADDR vgpr pair = 64-bit global address, GV mode (saddr = off).
    uint32_t lds_off = (uint32_t)(uintptr_t)lds_ptr;
    asm volatile("global_load_async_to_lds_b128 %0, %1, off"
                 :: "v"(lds_off), "v"((uint64_t)(uintptr_t)gptr)
                 : "memory");
}
__device__ __forceinline__ void wait_async0() {
    asm volatile("s_wait_asynccnt 0x0" ::: "memory");
}

// ---------------------------------------------------------------------------
// Prep: f16 shadows of W1 and W1^T, zero-init state buffers.
// ---------------------------------------------------------------------------
__global__ void prep_kernel(const float* __restrict__ W1,
                            _Float16* __restrict__ W116,
                            _Float16* __restrict__ W1T16,
                            float* __restrict__ h1f, float* __restrict__ h2f,
                            _Float16* __restrict__ h1h, _Float16* __restrict__ h2h) {
    for (int i = blockIdx.x * blockDim.x + threadIdx.x; i < HD * HD;
         i += gridDim.x * blockDim.x) {
        int n = i >> 10, k = i & 1023;
        W116[i]  = (_Float16)W1[i];
        W1T16[i] = (_Float16)W1[k * HD + n];
        h1f[i] = 0.f;  h2f[i] = 0.f;
        h1h[i] = (_Float16)0.f;  h2h[i] = (_Float16)0.f;
    }
}

// ---------------------------------------------------------------------------
// Phase 0: Afull = X @ W0^T + b0  via V_WMMA_F32_16X16X4_F32 (full f32).
// One-time 1.6 GFLOP; fragments read straight from global (L2 resident).
// ---------------------------------------------------------------------------
#if __has_builtin(__builtin_amdgcn_wmma_f32_16x16x4_f32)
#define HAVE_WMMA_F32 1
#else
#define HAVE_WMMA_F32 0
#endif

__global__ __launch_bounds__(128)
void gemm0_f32(const float* __restrict__ X,    // BATCH x DIN
               const float* __restrict__ W0,   // HD x DIN   (== B^T: row n, col k)
               const float* __restrict__ b0,   // HD
               float* __restrict__ Afull) {    // BATCH x HD
#if HAVE_WMMA_F32
    int lane   = threadIdx.x & 31;
    int wave   = threadIdx.x >> 5;
    int lane16 = lane & 15;
    int laneHi = lane >> 4;
    int bM = blockIdx.y * 64 + (wave >> 1) * 32;
    int bN = blockIdx.x * 64 + (wave & 1) * 32;

    v8f acc[2][2] = {};
    for (int k = 0; k < DIN; k += 4) {
        v2f a[2], b[2];
#pragma unroll
        for (int i = 0; i < 2; ++i)
            a[i] = *(const v2f*)(X + (size_t)(bM + i * 16 + lane16) * DIN + k + laneHi * 2);
#pragma unroll
        for (int j = 0; j < 2; ++j)
            b[j] = *(const v2f*)(W0 + (size_t)(bN + j * 16 + lane16) * DIN + k + laneHi * 2);
#pragma unroll
        for (int i = 0; i < 2; ++i)
#pragma unroll
            for (int j = 0; j < 2; ++j)
                acc[i][j] = __builtin_amdgcn_wmma_f32_16x16x4_f32(
                    false, a[i], false, b[j], (short)0, acc[i][j], false, false);
    }
#pragma unroll
    for (int i = 0; i < 2; ++i)
#pragma unroll
        for (int j = 0; j < 2; ++j)
#pragma unroll
            for (int r = 0; r < 8; ++r) {
                int row = bM + i * 16 + laneHi * 8 + r;
                int col = bN + j * 16 + lane16;
                Afull[(size_t)row * HD + col] = acc[i][j][r] + b0[col];
            }
#else
    int row = blockIdx.y * 64 + (threadIdx.x >> 5) * 16 + ((threadIdx.x & 31) & 15);
    int colBase = blockIdx.x * 64;
    for (int c = (threadIdx.x & 31) >> 4; c < 64; c += 2) {
        int col = colBase + c;
        float s = b0[col];
        for (int k = 0; k < DIN; ++k)
            s += X[(size_t)row * DIN + k] * W0[(size_t)col * DIN + k];
        Afull[(size_t)row * HD + col] = s;
    }
#endif
}

// ---------------------------------------------------------------------------
// Main GEMM: C = Amat(MxK,f16) * BT(NxK,f16)^T with fused eps-step epilogue.
//   out = beta*(acc + addmat? + bias?) + alpha*oldmat?   -> f32 (+ f16 shadow)
// WG = 4 waves / 128 thr; block tile 64(M) x 128(N); wave tile 32x64;
// K-step 32; double-buffered LDS filled by global_load_async_to_lds_b128.
// ---------------------------------------------------------------------------
#define LDS_STRIDE 40   // 32 halves + 8 pad: 80B rows => 16B aligned, bank-spread

__global__ __launch_bounds__(128)
void gemm_f16(const _Float16* __restrict__ Amat,  // M x K row-major
              const _Float16* __restrict__ BT,    // N x K row-major (B transposed)
              const float* __restrict__ oldmat,   // nullable, M x N
              const float* __restrict__ addmat,   // nullable, M x N
              const float* __restrict__ bias,     // nullable, N (indexed by col)
              float alpha, float beta,
              float* __restrict__ outF32,         // M x N
              _Float16* __restrict__ outF16) {    // nullable, M x N
    __shared__ __align__(16) _Float16 As[2][64 * LDS_STRIDE];
    __shared__ __align__(16) _Float16 Bs[2][128 * LDS_STRIDE];

    const int K = HD, N = HD;
    int tid    = threadIdx.x;
    int lane   = tid & 31;
    int wave   = tid >> 5;
    int lane16 = lane & 15;
    int laneHi = lane >> 4;
    int bM = blockIdx.y * 64;
    int bN = blockIdx.x * 128;
    int wM = (wave >> 1) * 32;   // 0 / 32
    int wN = (wave & 1) * 64;    // 0 / 64

    v8f acc[2][4] = {};

    // Stage one 32-wide K chunk of A (64 rows) and B^T (128 rows) into buf.
    // 16B chunks: A = 256 chunks (2/thread), B = 512 chunks (4/thread).
    auto stage = [&](int k0, int buf) {
#pragma unroll
        for (int c = tid; c < 64 * 4; c += 128) {
            int row = c >> 2, seg = c & 3;
            async_ldst_b128(&As[buf][row * LDS_STRIDE + seg * 8],
                            &Amat[(size_t)(bM + row) * K + k0 + seg * 8]);
        }
#pragma unroll
        for (int c = tid; c < 128 * 4; c += 128) {
            int row = c >> 2, seg = c & 3;
            async_ldst_b128(&Bs[buf][row * LDS_STRIDE + seg * 8],
                            &BT[(size_t)(bN + row) * K + k0 + seg * 8]);
        }
    };

    stage(0, 0);
    wait_async0();
    __syncthreads();

    int cur = 0;
    for (int k0 = 0; k0 < K; k0 += 32) {
        // Prefetch next chunk into the other buffer while we compute.
        if (k0 + 32 < K) stage(k0 + 32, cur ^ 1);

        // A frag (16x32 f16): halves 0-7 = K laneHi*8+0..7, halves 8-15 = +16.
        union HV { v16h v; uint4 q[2]; };
        HV af[2], bf[4];
#pragma unroll
        for (int i = 0; i < 2; ++i) {
            const _Float16* p = &As[cur][(wM + i * 16 + lane16) * LDS_STRIDE];
            af[i].q[0] = *(const uint4*)(p + laneHi * 8);
            af[i].q[1] = *(const uint4*)(p + 16 + laneHi * 8);
        }
        // B frag (32x16 f16): lane holds column n, halves 0-7 = K laneHi*8+0..7, halves 8-15 = +16.
#pragma unroll
        for (int j = 0; j < 4; ++j) {
            const _Float16* p = &Bs[cur][(wN + j * 16 + lane16) * LDS_STRIDE];
            bf[j].q[0] = *(const uint4*)(p + laneHi * 8);
            bf[j].q[1] = *(const uint4*)(p + 16 + laneHi * 8);
        }
#pragma unroll
        for (int i = 0; i < 2; ++i)
#pragma unroll
            for (int j = 0; j < 4; ++j)
                acc[i][j] = __builtin_amdgcn_wmma_f32_16x16x32_f16(
                    false, af[i].v, false, bf[j].v, (short)0, acc[i][j], false, false);

        // Wait for the prefetch, then rotate buffers at a full barrier.
        wait_async0();
        __syncthreads();
        cur ^= 1;
    }

    // Fused eps-step epilogue.
#pragma unroll
    for (int i = 0; i < 2; ++i)
#pragma unroll
        for (int j = 0; j < 4; ++j)
#pragma unroll
            for (int r = 0; r < 8; ++r) {
                int row = bM + wM + i * 16 + laneHi * 8 + r;
                int col = bN + wN + j * 16 + lane16;
                size_t idx = (size_t)row * N + col;
                float v = acc[i][j][r];
                if (addmat) v += addmat[idx];
                if (bias)   v += bias[col];
                v *= beta;
                if (oldmat) v += alpha * oldmat[idx];
                outF32[idx] = v;
                if (outF16) outF16[idx] = (_Float16)v;
            }
}

// ---------------------------------------------------------------------------
// Energy reduction: out[b] = sum_n [0.5h1^2 - h1*Afull + 0.5h2^2 - h2*P]
//                            - sum_k x[b,k]*vb[k]       (deterministic tree)
// ---------------------------------------------------------------------------
__global__ __launch_bounds__(256)
void energy_kernel(const float* __restrict__ X,
                   const float* __restrict__ vb,
                   const float* __restrict__ h1,
                   const float* __restrict__ h2,
                   const float* __restrict__ Afull,
                   const float* __restrict__ P,
                   float* __restrict__ out) {
    __shared__ float red[256];
    int b = blockIdx.x;
    int tid = threadIdx.x;
    float s = 0.f;
    for (int n = tid; n < HD; n += 256) {
        size_t idx = (size_t)b * HD + n;
        float a = h1[idx], c = h2[idx];
        s += 0.5f * a * a - a * Afull[idx];
        s += 0.5f * c * c - c * P[idx];
    }
    for (int k = tid; k < DIN; k += 256)
        s -= X[(size_t)b * DIN + k] * vb[k];
    red[tid] = s;
    __syncthreads();
    for (int st = 128; st > 0; st >>= 1) {
        if (tid < st) red[tid] += red[tid + st];
        __syncthreads();
    }
    if (tid == 0) out[b] = red[0];
}

// ---------------------------------------------------------------------------
// Host orchestration (graph-capture safe: only launches + pointer swaps).
// Workspace layout (32 MB):
//   [0,4)MB Afull | [4,12)MB h1f32 x2 | [12,20)MB h2f32 x2
//   [20,24)MB h1f16 x2 | [24,28)MB h2f16 x2 | [28,30)MB W1f16 | [30,32)MB W1Tf16
// ---------------------------------------------------------------------------
extern "C" void kernel_launch(void* const* d_in, const int* in_sizes, int n_in,
                              void* d_out, int out_size, void* d_ws, size_t ws_size,
                              hipStream_t stream) {
    const float* x  = (const float*)d_in[0];
    const float* vb = (const float*)d_in[1];
    const float* W0 = (const float*)d_in[2];
    const float* b0 = (const float*)d_in[3];
    const float* W1 = (const float*)d_in[4];
    const float* b1 = (const float*)d_in[5];
    float* out = (float*)d_out;

    char* ws = (char*)d_ws;
    const size_t MB = 1u << 20;
    float*    Afull  = (float*)(ws + 0 * MB);
    float*    h1f[2] = { (float*)(ws + 4 * MB),  (float*)(ws + 8 * MB)  };
    float*    h2f[2] = { (float*)(ws + 12 * MB), (float*)(ws + 16 * MB) };
    _Float16* h1h[2] = { (_Float16*)(ws + 20 * MB), (_Float16*)(ws + 22 * MB) };
    _Float16* h2h[2] = { (_Float16*)(ws + 24 * MB), (_Float16*)(ws + 26 * MB) };
    _Float16* W116   = (_Float16*)(ws + 28 * MB);
    _Float16* W1T16  = (_Float16*)(ws + 30 * MB);

    prep_kernel<<<1024, 256, 0, stream>>>(W1, W116, W1T16,
                                          h1f[0], h2f[0], h1h[0], h2h[0]);

    gemm0_f32<<<dim3(16, 16), 128, 0, stream>>>(x, W0, b0, Afull);

    dim3 gg(HD / 128, BATCH / 64);   // (8, 16)
    int cur = 0;
    for (int t = 0; t < TSTEPS; ++t) {
        int nxt = cur ^ 1;
        // h1' = 0.5*h1 + 0.5*(Afull + h2 @ W1):  B^T = W1^T
        gemm_f16<<<gg, 128, 0, stream>>>(h2h[cur], W1T16,
                                         h1f[cur], Afull, nullptr,
                                         0.5f, 0.5f, h1f[nxt], h1h[nxt]);
        // h2' = 0.5*h2 + 0.5*(h1 @ W1^T + b1):  B^T = W1
        gemm_f16<<<gg, 128, 0, stream>>>(h1h[cur], W116,
                                         h2f[cur], nullptr, b1,
                                         0.5f, 0.5f, h2f[nxt], h2h[nxt]);
        cur = nxt;
    }
    // TSTEPS even -> cur == 0, buffer 1 is free: use h1f[1] as P scratch.
    float* P = h1f[1];
    gemm_f16<<<gg, 128, 0, stream>>>(h1h[cur], W116,
                                     nullptr, nullptr, b1,
                                     0.0f, 1.0f, P, nullptr);

    energy_kernel<<<BATCH, 256, 0, stream>>>(x, vb, h1f[cur], h2f[cur],
                                             Afull, P, out);
}
